// MultiHeadSelfAttention_40673340293890
// MI455X (gfx1250) — matrix operand order repaired and checked by reading, not compiled, not run on hardware
//
#include <hip/hip_runtime.h>
#include <hip/hip_bf16.h>
#include <cstdint>

#define DEV __device__ __forceinline__

typedef __bf16 bf16_t;
typedef __attribute__((ext_vector_type(16))) __bf16 v16bf;
typedef __attribute__((ext_vector_type(8)))  __bf16 v8bf;
typedef __attribute__((ext_vector_type(8)))  float  v8f;

constexpr int BB = 4;        // batch
constexpr int T  = 2048;     // seq len
constexpr int D  = 1024;     // model dim
constexpr int H  = 16;       // heads
constexpr int HD = 64;       // head dim
constexpr int BT = BB * T;   // 8192 rows

DEV v16bf cat8(v8bf lo, v8bf hi) {
  return __builtin_shufflevector(lo, hi, 0,1,2,3,4,5,6,7,8,9,10,11,12,13,14,15);
}
DEV v8bf ld8(const bf16_t* p) { return *reinterpret_cast<const v8bf*>(p); }
DEV v8f wmma_bf(v16bf a, v16bf b, v8f c) {
  // D = A(16x32 bf16) x B(32x16 bf16) + C(16x16 f32)
  return __builtin_amdgcn_wmma_f32_16x16x32_bf16(false, a, false, b, (short)0, c, false, false);
}
DEV v8f vzero() { v8f z = {0.f,0.f,0.f,0.f,0.f,0.f,0.f,0.f}; return z; }

// ---------------------------------------------------------------------------
// Kernel 1: fp32 -> bf16 conversion
// ---------------------------------------------------------------------------
__global__ void cvt_kernel(const float* __restrict__ src, bf16_t* __restrict__ dst, int n) {
  int i = blockIdx.x * blockDim.x + threadIdx.x;
  if (i < n) dst[i] = (bf16_t)src[i];
}

// ---------------------------------------------------------------------------
// Kernel 2: fused QKV projection + bias + RoPE (q,k).
// grid = (BT/64, H, 3), block = 128 (4 waves). Wave: 16 rows x 64 cols (one head).
// A fragment (16x32, 16-bit): lane row m = ln; elem i -> k = i + 16*hi        -> two b128 loads
// B fragment (32x16, 16-bit): lane col n = ln; elem i -> k = i + 16*hi        -> two b128 loads (W is [out,in])
// C/D layout: lane -> N = ln, reg r -> M = r + 8*hi
// ---------------------------------------------------------------------------
__global__ __launch_bounds__(128) void qkv_rope_kernel(
    const bf16_t* __restrict__ X, const bf16_t* __restrict__ Wall,
    const float* __restrict__ bqp, const float* __restrict__ bkp, const float* __restrict__ bvp,
    bf16_t* __restrict__ Qb, bf16_t* __restrict__ Kb, bf16_t* __restrict__ Vt)
{
  const int lane = threadIdx.x & 31;
  const int wave = threadIdx.x >> 5;
  const int ln = lane & 15;
  const int hi = lane >> 4;
  const int mat = blockIdx.z;                 // 0=Q 1=K 2=V
  const int h   = blockIdx.y;
  const int m0  = (blockIdx.x * 4 + wave) * 16;
  const int n0  = h * HD;
  const bf16_t* W = Wall + (size_t)mat * D * D;
  const float* bias = (mat == 0) ? bqp : ((mat == 1) ? bkp : bvp);

  v8f acc[4] = { vzero(), vzero(), vzero(), vzero() };
  const bf16_t* arow = X + (size_t)(m0 + ln) * D;

  for (int k0 = 0; k0 < D; k0 += 32) {
    v16bf af = cat8(ld8(arow + k0 + hi * 16), ld8(arow + k0 + hi * 16 + 8));
    #pragma unroll
    for (int a = 0; a < 4; a++) {
      const bf16_t* wp = W + (size_t)(n0 + a * 16 + ln) * D + k0 + hi * 16;
      v16bf wf = cat8(ld8(wp), ld8(wp + 8));
      acc[a] = wmma_bf(af, wf, acc[a]);
    }
  }
  #pragma unroll
  for (int a = 0; a < 4; a++) {
    float bval = bias[n0 + a * 16 + ln];
    #pragma unroll
    for (int r = 0; r < 8; r++) acc[a][r] += bval;
  }

  if (mat < 2) {
    // RoPE: pair j (acc[a]) with j+32 (acc[a+2]); angle = t * 10000^(-j/32), j = a*16+ln < 32
    #pragma unroll
    for (int a = 0; a < 2; a++) {
      float fi  = (float)(a * 16 + ln);
      float inv = __expf(fi * -0.28782313662425572f);   // ln(10000)/32
      #pragma unroll
      for (int r = 0; r < 8; r++) {
        int   m   = m0 + r + hi * 8;
        float t   = (float)(m & (T - 1));
        float ang = t * inv;
        float c = __cosf(ang), s = __sinf(ang);
        float x1 = acc[a][r], x2 = acc[a + 2][r];
        acc[a][r]     = x1 * c - x2 * s;
        acc[a + 2][r] = x2 * c + x1 * s;
      }
    }
    bf16_t* dst = (mat == 0) ? Qb : Kb;          // [B,H,T,hd]
    #pragma unroll
    for (int a = 0; a < 4; a++)
      #pragma unroll
      for (int r = 0; r < 8; r++) {
        int m = m0 + r + hi * 8;
        int b = m >> 11, t = m & (T - 1);
        dst[((((size_t)b * H + h) * T + t) << 6) + a * 16 + ln] = (bf16_t)acc[a][r];
      }
  } else {
    // V stored transposed: [B,H,hd,T]
    #pragma unroll
    for (int a = 0; a < 4; a++)
      #pragma unroll
      for (int r = 0; r < 8; r++) {
        int m = m0 + r + hi * 8;
        int b = m >> 11, t = m & (T - 1);
        Vt[(((size_t)b * H + h) * HD + a * 16 + ln) * (size_t)T + t] = (bf16_t)acc[a][r];
      }
  }
}

// ---------------------------------------------------------------------------
// Kernel 3: causal flash attention, computed transposed.
//   S^T = K·Q^T  (C layout: lane -> query=ln, reg r -> key = r+8*hi (+16 per tile))
//   online softmax per query = within-lane reduce + one shfl_xor(16)
//   O^T = V^T · P^T, accumulated over 32-key chunks (one K=32 WMMA per 16-d strip)
// grid = (T/64, B*H), block = 128; each wave owns 16 queries.
// ---------------------------------------------------------------------------
__global__ __launch_bounds__(128) void flash_attn_kernel(
    const bf16_t* __restrict__ Qb, const bf16_t* __restrict__ Kb,
    const bf16_t* __restrict__ Vt, bf16_t* __restrict__ AO)
{
  const int lane = threadIdx.x & 31;
  const int wave = threadIdx.x >> 5;
  const int ln = lane & 15, hi = lane >> 4;
  const int bh = blockIdx.y;
  const int b = bh >> 4, h = bh & 15;
  const int qbase = blockIdx.x * 64 + wave * 16;

  const bf16_t* Qh = Qb + (size_t)bh * T * HD;
  const bf16_t* Kh = Kb + (size_t)bh * T * HD;
  const bf16_t* Vh = Vt + (size_t)bh * HD * T;

  // Q as B-fragments (elem i -> d = (i<8?i:i+8) + 8*hi, +32 for second half of hd)
  const bf16_t* qrow = Qh + (size_t)(qbase + ln) * HD + hi * 8;
  v16bf qf0 = cat8(ld8(qrow),      ld8(qrow + 16));
  v16bf qf1 = cat8(ld8(qrow + 32), ld8(qrow + 48));

  v8f oacc[4] = { vzero(), vzero(), vzero(), vzero() };
  float mrun = -1e30f, lrun = 0.f;

  const int nch = qbase / 32 + 1;
  for (int c = 0; c < nch; c++) {
    const int kb = c * 32;
    v8f s0, s1;
    {
      const bf16_t* kr0 = Kh + (size_t)(kb + ln) * HD + hi * 8;
      v8f z = vzero();
      z  = wmma_bf(cat8(ld8(kr0),      ld8(kr0 + 16)), qf0, z);
      s0 = wmma_bf(cat8(ld8(kr0 + 32), ld8(kr0 + 48)), qf1, z);
      const bf16_t* kr1 = Kh + (size_t)(kb + 16 + ln) * HD + hi * 8;
      v8f z2 = vzero();
      z2 = wmma_bf(cat8(ld8(kr1),      ld8(kr1 + 16)), qf0, z2);
      s1 = wmma_bf(cat8(ld8(kr1 + 32), ld8(kr1 + 48)), qf1, z2);
    }
    const float scale = 0.125f;                 // 1/sqrt(64)
    const int q = qbase + ln;
    const bool needmask = (kb + 31 > qbase);
    #pragma unroll
    for (int r = 0; r < 8; r++) {
      int key0 = kb + r + hi * 8;
      float v0 = s0[r] * scale;
      float v1 = s1[r] * scale;
      if (needmask) {
        if (key0 > q)      v0 = -1e30f;
        if (key0 + 16 > q) v1 = -1e30f;
      }
      s0[r] = v0; s1[r] = v1;
    }
    float cmax = -1e30f;
    #pragma unroll
    for (int r = 0; r < 8; r++) cmax = fmaxf(cmax, fmaxf(s0[r], s1[r]));
    cmax = fmaxf(cmax, __shfl_xor(cmax, 16, 32));
    float mnew = fmaxf(mrun, cmax);
    float corr = __expf(mrun - mnew);
    float rsum = 0.f;
    #pragma unroll
    for (int r = 0; r < 8; r++) {
      float p0 = __expf(s0[r] - mnew), p1 = __expf(s1[r] - mnew);
      s0[r] = p0; s1[r] = p1; rsum += p0 + p1;
    }
    rsum += __shfl_xor(rsum, 16, 32);
    lrun = lrun * corr + rsum;
    mrun = mnew;
    #pragma unroll
    for (int a = 0; a < 4; a++)
      #pragma unroll
      for (int r = 0; r < 8; r++) oacc[a][r] *= corr;

    // Build P^T B-fragment: elem i -> key = kb + i + 16*hi, col = ln.
    // half0 owns i<8 in s0, needs partner's s0 for i>=8; half1 symmetric with s1.
    v16bf pf;
    #pragma unroll
    for (int r = 0; r < 8; r++) {
      float send = hi ? s0[r] : s1[r];
      float recv = __shfl_xor(send, 16, 32);
      float elo  = hi ? recv  : s0[r];
      float ehi  = hi ? s1[r] : recv;
      pf[r]     = (bf16_t)elo;
      pf[r + 8] = (bf16_t)ehi;
    }
    #pragma unroll
    for (int a = 0; a < 4; a++) {
      const bf16_t* vp = Vh + (size_t)(a * 16 + ln) * T + kb + hi * 16;
      oacc[a] = wmma_bf(cat8(ld8(vp), ld8(vp + 8)), pf, oacc[a]);
    }
  }

  float inv = 1.0f / lrun;
  // O^T C-layout: lane -> query = ln, M = d = 16a + 8*hi + r  -> packed b128 stores
  bf16_t* orow = AO + (size_t)(b * T + qbase + ln) * D + h * HD + hi * 8;
  #pragma unroll
  for (int a = 0; a < 4; a++) {
    v8bf pk;
    #pragma unroll
    for (int r = 0; r < 8; r++) pk[r] = (bf16_t)(oacc[a][r] * inv);
    *reinterpret_cast<v8bf*>(orow + a * 16) = pk;
  }
}

// ---------------------------------------------------------------------------
// Kernel 4: output projection  out = AO·Wo^T + bo  (fp32 out)
// grid = (BT/64, D/64), block = 128
// ---------------------------------------------------------------------------
__global__ __launch_bounds__(128) void oproj_kernel(
    const bf16_t* __restrict__ A, const bf16_t* __restrict__ Wo,
    const float* __restrict__ bo, float* __restrict__ out)
{
  const int lane = threadIdx.x & 31;
  const int wave = threadIdx.x >> 5;
  const int ln = lane & 15, hi = lane >> 4;
  const int m0 = (blockIdx.x * 4 + wave) * 16;
  const int n0 = blockIdx.y * 64;

  v8f acc[4] = { vzero(), vzero(), vzero(), vzero() };
  const bf16_t* arow = A + (size_t)(m0 + ln) * D;
  for (int k0 = 0; k0 < D; k0 += 32) {
    v16bf af = cat8(ld8(arow + k0 + hi * 16), ld8(arow + k0 + hi * 16 + 8));
    #pragma unroll
    for (int a = 0; a < 4; a++) {
      const bf16_t* wp = Wo + (size_t)(n0 + a * 16 + ln) * D + k0 + hi * 16;
      acc[a] = wmma_bf(af, cat8(ld8(wp), ld8(wp + 8)), acc[a]);
    }
  }
  #pragma unroll
  for (int a = 0; a < 4; a++) {
    float bval = bo[n0 + a * 16 + ln];
    #pragma unroll
    for (int r = 0; r < 8; r++)
      out[(size_t)(m0 + r + hi * 8) * D + n0 + a * 16 + ln] = acc[a][r] + bval;
  }
}

// ---------------------------------------------------------------------------
extern "C" void kernel_launch(void* const* d_in, const int* in_sizes, int n_in,
                              void* d_out, int out_size, void* d_ws, size_t ws_size,
                              hipStream_t stream) {
  (void)in_sizes; (void)n_in; (void)out_size; (void)ws_size;
  const float* x  = (const float*)d_in[0];
  const float* Wq = (const float*)d_in[1];
  const float* bq = (const float*)d_in[2];
  const float* Wk = (const float*)d_in[3];
  const float* bk = (const float*)d_in[4];
  const float* Wv = (const float*)d_in[5];
  const float* bv = (const float*)d_in[6];
  const float* Wo = (const float*)d_in[7];
  const float* bo = (const float*)d_in[8];
  float* out = (float*)d_out;

  char* ws = (char*)d_ws;
  const size_t XBF = 0;                              // x bf16 (reused as attn-out)
  const size_t WBF = (size_t)BT * D * 2;             // 4 weight mats bf16
  const size_t QB  = WBF + (size_t)4 * D * D * 2;    // Q [B,H,T,hd]
  const size_t KB  = QB + (size_t)BT * D * 2;        // K [B,H,T,hd]
  const size_t VT  = KB + (size_t)BT * D * 2;        // V^T [B,H,hd,T]
  bf16_t* xbf = (bf16_t*)(ws + XBF);
  bf16_t* wbf = (bf16_t*)(ws + WBF);
  bf16_t* qb  = (bf16_t*)(ws + QB);
  bf16_t* kb  = (bf16_t*)(ws + KB);
  bf16_t* vt  = (bf16_t*)(ws + VT);
  bf16_t* ao  = (bf16_t*)(ws + XBF);                 // reuse xbf region after QKV

  const int CV = 256;
  cvt_kernel<<<(BT * D + CV - 1) / CV, CV, 0, stream>>>(x, xbf, BT * D);
  cvt_kernel<<<(D * D + CV - 1) / CV, CV, 0, stream>>>(Wq, wbf + (size_t)0 * D * D, D * D);
  cvt_kernel<<<(D * D + CV - 1) / CV, CV, 0, stream>>>(Wk, wbf + (size_t)1 * D * D, D * D);
  cvt_kernel<<<(D * D + CV - 1) / CV, CV, 0, stream>>>(Wv, wbf + (size_t)2 * D * D, D * D);
  cvt_kernel<<<(D * D + CV - 1) / CV, CV, 0, stream>>>(Wo, wbf + (size_t)3 * D * D, D * D);

  qkv_rope_kernel<<<dim3(BT / 64, H, 3), 128, 0, stream>>>(xbf, wbf, bq, bk, bv, qb, kb, vt);
  flash_attn_kernel<<<dim3(T / 64, BB * H), 128, 0, stream>>>(qb, kb, vt, ao);
  oproj_kernel<<<dim3(BT / 64, D / 64), 128, 0, stream>>>(ao, wbf + (size_t)3 * D * D, bo, out);
}
